// MultiHeadedAttention_66073776882043
// MI455X (gfx1250) — matrix operand order repaired and checked by reading, not compiled, not run on hardware
//
#include <hip/hip_runtime.h>

// ---------------------------------------------------------------------------
// MHA with log_softmax weights, decomposed as:
//   x = (S @ V) - lse ⊗ colsum(V),   S = mask(QK^T / sqrt(dk))
// All GEMMs run on v_wmma_f32_16x16x32_bf16 (fp32 accumulate).
// LDS fragment loads are contiguous (ds_load_b128); global stores are
// coalesced b128 via LDS writeback tiles; projection K-loop double-buffered;
// attention tile staging uses async global->LDS copies when available.
// ---------------------------------------------------------------------------

typedef __attribute__((ext_vector_type(16))) __bf16          v16bf;
typedef __attribute__((ext_vector_type(2)))  __bf16          v2bf;
typedef __attribute__((ext_vector_type(16))) unsigned short  v16u;
typedef __attribute__((ext_vector_type(8)))  float           v8f;

constexpr int kB  = 2;
constexpr int kH  = 16;
constexpr int kS  = 2048;
constexpr int kD  = 1024;   // d_model
constexpr int kDk = 64;     // per-head dim
constexpr float kNeg = -1000000000.0f;

#if defined(__gfx1250__) &&                                             \
    __has_builtin(__builtin_amdgcn_global_load_async_to_lds_b128) &&    \
    __has_builtin(__builtin_amdgcn_s_wait_asynccnt)
#define USE_ASYNC_LDS 1
#else
#define USE_ASYNC_LDS 0
#endif

#if USE_ASYNC_LDS
typedef int v4i_vs __attribute__((vector_size(16)));
__device__ __forceinline__ void async_copy16(void* lds_dst, const void* gsrc) {
  __builtin_amdgcn_global_load_async_to_lds_b128(
      (v4i_vs*)gsrc, (v4i_vs*)lds_dst, 0, 0);
}
#endif

__device__ __forceinline__ unsigned pack2bf(float lo, float hi) {
#if __has_builtin(__builtin_amdgcn_cvt_pk_bf16_f32)
  v2bf r = __builtin_amdgcn_cvt_pk_bf16_f32(lo, hi);
  return __builtin_bit_cast(unsigned, r);
#else
  union { float f; unsigned u; } a, b; a.f = lo; b.f = hi;
  unsigned ua = (a.u + 0x7FFFu + ((a.u >> 16) & 1u)) >> 16;
  unsigned ub = (b.u + 0x7FFFu + ((b.u >> 16) & 1u)) & 0xFFFF0000u;
  return ua | ub;
#endif
}
__device__ __forceinline__ unsigned short f2bf(float f) {
  return (unsigned short)pack2bf(f, f);
}
__device__ __forceinline__ float bf2f(unsigned short h) {
  union { unsigned u; float f; } c; c.u = ((unsigned)h) << 16;
  return c.f;
}
__device__ __forceinline__ uint4 pack8(float4 a, float4 b) {
  uint4 p;
  p.x = pack2bf(a.x, a.y); p.y = pack2bf(a.z, a.w);
  p.z = pack2bf(b.x, b.y); p.w = pack2bf(b.z, b.w);
  return p;
}

__device__ __forceinline__ v8f wmma_bf16(v16u a, v16u b, v8f c) {
  return __builtin_amdgcn_wmma_f32_16x16x32_bf16(
      false, __builtin_bit_cast(v16bf, a),
      false, __builtin_bit_cast(v16bf, b),
      (short)0, c, false, false);
}

// A-matrix 16x32 bf16 fragment (CDNA5 layout): two contiguous 16B runs/lane.
__device__ __forceinline__ v16u load_a(const unsigned short* sm, int stride,
                                       int m0, int k0, int lane) {
  const int half = lane >> 4;
  const unsigned short* row = sm + (m0 + (lane & 15)) * stride;
  v16u v;
#pragma unroll
  for (int e = 0; e < 16; ++e) {
    int k = k0 + e + ((e < 8) ? 0 : 8) + half * 8;
    v[e] = row[k];
  }
  return v;
}

// B-matrix 32x16 bf16 fragment from an [N][K]-stored LDS tile: two 16B runs/lane.
__device__ __forceinline__ v16u load_bT(const unsigned short* sm, int stride,
                                        int k0, int n0, int lane) {
  const int half = lane >> 4;
  const unsigned short* row = sm + (n0 + (lane & 15)) * stride;
  v16u v;
#pragma unroll
  for (int e = 0; e < 16; ++e) v[e] = row[k0 + e + ((e < 8) ? 0 : 8) + half * 8];
  return v;
}

// ---------------------------------------------------------------------------
// out = A[4096,1024] @ W[1024,1024]^T + bias
//   MODE 0: bf16 head-major   [B,H,S,64]   (Q, K)
//   MODE 2: bf16 head-transp. [B,H,64,S]   (V, pre-transposed for S@V)
//   MODE 1: fp32 row-major    [4096,1024]  (final output)
// ---------------------------------------------------------------------------
template <int MODE>
__global__ __launch_bounds__(256) void proj_gemm_kernel(
    const float* __restrict__ A, const float* __restrict__ W,
    const float* __restrict__ bias, unsigned short* __restrict__ outb,
    float* __restrict__ outf) {
  constexpr int SA = 40, SB = 40;  // padded LDS strides (elements)
  __shared__ unsigned short As[2][128 * SA];  // [m][k] bf16
  __shared__ unsigned short Bs[2][64 * SB];   // [n][k] bf16 (W row-major slice)
  __shared__ float Cb[8 * 32 * 32];           // per-wave writeback tile (fp32)

  const int tid  = threadIdx.x;
  const int lane = tid & 31, w = tid >> 5;
  const int m0b = blockIdx.y * 128, n0b = blockIdx.x * 64;
  const int mw = 32 * (w >> 1), nw = 32 * (w & 1);

  auto stage = [&](int pb, int k0) {
#pragma unroll
    for (int i = 0; i < 4; ++i) {            // A 128x32 -> bf16
      int idx = tid + 256 * i;
      int r = idx >> 3, c4 = (idx & 7) * 4;
      const float4 f = *(const float4*)&A[(size_t)(m0b + r) * kD + k0 + c4];
      uint2 p; p.x = pack2bf(f.x, f.y); p.y = pack2bf(f.z, f.w);
      *(uint2*)&As[pb][r * SA + c4] = p;
    }
#pragma unroll
    for (int i = 0; i < 2; ++i) {            // B 64x32 -> bf16 (n-major)
      int idx = tid + 256 * i;
      int n = idx >> 3, c4 = (idx & 7) * 4;
      const float4 f = *(const float4*)&W[(size_t)(n0b + n) * kD + k0 + c4];
      uint2 p; p.x = pack2bf(f.x, f.y); p.y = pack2bf(f.z, f.w);
      *(uint2*)&Bs[pb][n * SB + c4] = p;
    }
  };

  v8f acc[2][2];
#pragma unroll
  for (int i = 0; i < 2; ++i)
#pragma unroll
    for (int j = 0; j < 2; ++j) acc[i][j] = (v8f){0,0,0,0,0,0,0,0};

  stage(0, 0);
  int pb = 0;
  for (int k0 = 0; k0 < kD; k0 += 32) {
    __syncthreads();
    if (k0 + 32 < kD) stage(pb ^ 1, k0 + 32);  // overlap next stage with WMMA
    v16u af0 = load_a (As[pb], SA, mw +  0, 0, lane);
    v16u af1 = load_a (As[pb], SA, mw + 16, 0, lane);
    v16u bf0 = load_bT(Bs[pb], SB, 0, nw +  0, lane);
    v16u bf1 = load_bT(Bs[pb], SB, 0, nw + 16, lane);
    acc[0][0] = wmma_bf16(af0, bf0, acc[0][0]);
    acc[0][1] = wmma_bf16(af0, bf1, acc[0][1]);
    acc[1][0] = wmma_bf16(af1, bf0, acc[1][0]);
    acc[1][1] = wmma_bf16(af1, bf1, acc[1][1]);
    pb ^= 1;
  }

  // ---- writeback: scatter (+bias) into per-wave LDS tile, gather as b128 ----
  const int hl = lane >> 4, nl = lane & 15;
  float* Cw = &Cb[w * 1024];
  const float bias0 = bias[n0b + nw + nl];
  const float bias1 = bias[n0b + nw + 16 + nl];
#pragma unroll
  for (int i = 0; i < 2; ++i)
#pragma unroll
    for (int j = 0; j < 2; ++j) {
      float bj = j ? bias1 : bias0;
#pragma unroll
      for (int r = 0; r < 8; ++r) {
        int lr = 16 * i + r + 8 * hl, lc = 16 * j + nl;
        float val = acc[i][j][r] + bj;
        if (MODE == 2) Cw[lc * 32 + lr] = val;   // transposed for [d][s] out
        else           Cw[lr * 32 + lc] = val;
      }
    }
  __syncthreads();

  const int gr = lane >> 2, g8 = (lane & 3) * 8;
#pragma unroll
  for (int c = 0; c < 4; ++c) {
    int lr2 = gr + 8 * c;
    float4 f0 = *(float4*)&Cw[lr2 * 32 + g8];
    float4 f1 = *(float4*)&Cw[lr2 * 32 + g8 + 4];
    if (MODE == 1) {
      int row = m0b + mw + lr2;
      *(float4*)&outf[(size_t)row * kD + n0b + nw + g8] = f0;
      *(float4*)&outf[(size_t)row * kD + n0b + nw + g8 + 4] = f1;
    } else if (MODE == 0) {
      int row = m0b + mw + lr2;          // s
      int col = n0b + nw + g8;           // d base
      int bb = row >> 11, s = row & 2047, hh = col >> 6, d = col & 63;
      *(uint4*)&outb[((size_t)(bb * kH + hh) * kS + s) * kDk + d] = pack8(f0, f1);
    } else {                             // MODE 2: Cw transposed, contiguous s
      int col = n0b + nw + lr2;          // global col -> (h, d)
      int row = m0b + mw + g8;           // s base
      int bb = row >> 11, s = row & 2047, hh = col >> 6, d = col & 63;
      *(uint4*)&outb[((size_t)(bb * kH + hh) * kDk + d) * kS + s] = pack8(f0, f1);
    }
  }
}

// ---------------------------------------------------------------------------
// Streaming attention: one block per (b, h, 64-row q tile).
//   acc = Σ_tiles mask(QK^T/8) @ V  (raw scores, bf16 WMMA)
//   lse via online max/sumexp in LDS;  vcol = Σ_k V[k,:]
//   X = acc - lse ⊗ vcol, written fp32 [4096, 1024] (heads concatenated).
// V arrives pre-transposed [B,H,64(d),S]; all LDS reads contiguous.
// Tile staging via GLOBAL_LOAD_ASYNC_TO_LDS_B128 (ASYNCcnt) when available.
// ---------------------------------------------------------------------------
__global__ __launch_bounds__(256) void attn_kernel(
    const unsigned short* __restrict__ Qb, const unsigned short* __restrict__ Kb,
    const unsigned short* __restrict__ Vt, const int* __restrict__ mask,
    float* __restrict__ X) {
  __shared__ unsigned short Qt[64 * 64];    // [q][d]   bf16
  __shared__ unsigned short Kt[64 * 64];    // [key][d] bf16
  __shared__ unsigned short VtT[64 * 64];   // [d][key] bf16
  __shared__ float          Sf[64 * 64];    // fp32 scores / X writeback tile
  __shared__ unsigned short Sb[64 * 64];    // bf16 scores for S@V
  __shared__ float s_rowm[64], s_rowl[64], s_newm[64], s_lse[64];
  __shared__ float s_part[64 * 4];
  __shared__ float s_vpart[64 * 4], s_vcol[64];

  const int tid = threadIdx.x, lane = tid & 31, w = tid >> 5;
  const int qt = blockIdx.x, h = blockIdx.y, b = blockIdx.z;
  const int mrow0 = 16 * (w & 3);   // q rows owned by this wave
  const int ncol0 = 32 * (w >> 2);  // key / d columns owned by this wave
  const int hl = lane >> 4, nl = lane & 15;
  const int srow = tid >> 2, sch = tid & 3;  // stats / vcol mapping

  const size_t headbase = (size_t)(b * kH + h) * kS * kDk;

  {  // stage Q tile: 64x64 bf16 = 512 x 16B
    const uint4* g = (const uint4*)(Qb + headbase + (size_t)qt * 64 * kDk);
    uint4* sq = (uint4*)Qt;
#if USE_ASYNC_LDS
#pragma unroll
    for (int i = 0; i < 2; ++i) async_copy16(&sq[tid + 256 * i], &g[tid + 256 * i]);
    __builtin_amdgcn_s_wait_asynccnt(0);
#else
#pragma unroll
    for (int i = 0; i < 2; ++i) sq[tid + 256 * i] = g[tid + 256 * i];
#endif
  }
  if (tid < 64) { s_rowm[tid] = -INFINITY; s_rowl[tid] = 0.f; }
  __syncthreads();

  v16u qa0 = load_a(Qt, 64, mrow0, 0, lane);
  v16u qa1 = load_a(Qt, 64, mrow0, 32, lane);

  v8f acc[2];
  acc[0] = (v8f){0,0,0,0,0,0,0,0};
  acc[1] = (v8f){0,0,0,0,0,0,0,0};
  float vacc = 0.f;

  for (int kt = 0; kt < kS / 64; ++kt) {
    __syncthreads();
    {  // stage K tile [key][d] and V^T tile [d][key]
      const uint4* gk = (const uint4*)(Kb + headbase + (size_t)kt * 64 * kDk);
      uint4* sk = (uint4*)Kt;
#if USE_ASYNC_LDS
#pragma unroll
      for (int i = 0; i < 2; ++i)
        async_copy16(&sk[tid + 256 * i], &gk[tid + 256 * i]);
#pragma unroll
      for (int i = 0; i < 2; ++i) {
        int u = tid + 256 * i;                 // 512 x 16B: [d][8-key chunk]
        int d = u >> 3, k8 = (u & 7) * 8;
        async_copy16(&VtT[d * 64 + k8],
                     &Vt[headbase + (size_t)d * kS + kt * 64 + k8]);
      }
      __builtin_amdgcn_s_wait_asynccnt(0);
#else
#pragma unroll
      for (int i = 0; i < 2; ++i) sk[tid + 256 * i] = gk[tid + 256 * i];
#pragma unroll
      for (int i = 0; i < 2; ++i) {
        int u = tid + 256 * i;                 // 512 uint4: [d][8-key chunk]
        int d = u >> 3, k8 = (u & 7) * 8;
        *(uint4*)&VtT[d * 64 + k8] =
            *(const uint4*)&Vt[headbase + (size_t)d * kS + kt * 64 + k8];
      }
#endif
    }
    if (kt + 1 < kS / 64) {  // prefetch next tiles into the cache hierarchy
      __builtin_prefetch(Kb + headbase + (size_t)(kt + 1) * 64 * kDk + tid * 16, 0, 1);
      __builtin_prefetch(Vt + headbase + (size_t)(tid & 63) * kS + (kt + 1) * 64, 0, 1);
    }
    __syncthreads();

    // ---- S tile = Q @ K^T * 0.125, mask, stage to LDS (fp32 + bf16) ----
#pragma unroll
    for (int tj = 0; tj < 2; ++tj) {
      v8f sc = (v8f){0,0,0,0,0,0,0,0};
      v16u b0 = load_bT(Kt, 64, 0,  ncol0 + 16 * tj, lane);
      v16u b1 = load_bT(Kt, 64, 32, ncol0 + 16 * tj, lane);
      sc = wmma_bf16(qa0, b0, sc);
      sc = wmma_bf16(qa1, b1, sc);
#pragma unroll
      for (int r = 0; r < 8; ++r) {
        int ql = mrow0 + r + 8 * hl;
        int kl = ncol0 + 16 * tj + nl;
        float val = sc[r] * 0.125f;
        int qg = qt * 64 + ql, kg = kt * 64 + kl;
        if (mask[((size_t)b * kS + qg) * kS + kg] == 0) val = kNeg;
        Sf[ql * 64 + kl] = val;
        Sb[ql * 64 + kl] = f2bf(val);
      }
    }
    __syncthreads();

    // ---- online logsumexp: per-row max then sum of exp ----
    {
      float mx = -INFINITY;
#pragma unroll
      for (int j = 0; j < 16; ++j) mx = fmaxf(mx, Sf[srow * 64 + sch * 16 + j]);
      s_part[srow * 4 + sch] = mx;
    }
    __syncthreads();
    if (tid < 64) {
      float m = fmaxf(fmaxf(s_part[tid * 4], s_part[tid * 4 + 1]),
                      fmaxf(s_part[tid * 4 + 2], s_part[tid * 4 + 3]));
      s_newm[tid] = fmaxf(s_rowm[tid], m);
    }
    __syncthreads();
    {
      float nm = s_newm[srow], sm = 0.f;
#pragma unroll
      for (int j = 0; j < 16; ++j) sm += __expf(Sf[srow * 64 + sch * 16 + j] - nm);
      s_part[srow * 4 + sch] = sm;
    }
    __syncthreads();
    if (tid < 64) {
      float nm = s_newm[tid];
      s_rowl[tid] = s_rowl[tid] * __expf(s_rowm[tid] - nm) + s_part[tid * 4] +
                    s_part[tid * 4 + 1] + s_part[tid * 4 + 2] + s_part[tid * 4 + 3];
      s_rowm[tid] = nm;
    }

    // ---- V column partial sums (contiguous reads of V^T row) ----
#pragma unroll
    for (int j = 0; j < 16; ++j) vacc += bf2f(VtT[srow * 64 + sch * 16 + j]);

    // ---- acc += S @ V ----
#pragma unroll
    for (int kk = 0; kk < 2; ++kk) {
      v16u af = load_a(Sb, 64, mrow0, kk * 32, lane);
#pragma unroll
      for (int tj = 0; tj < 2; ++tj) {
        v16u bfr = load_bT(VtT, 64, kk * 32, ncol0 + 16 * tj, lane);
        acc[tj] = wmma_bf16(af, bfr, acc[tj]);
      }
    }
  }

  __syncthreads();
  if (tid < 64) s_lse[tid] = s_rowm[tid] + __logf(s_rowl[tid]);
  s_vpart[srow * 4 + sch] = vacc;
  __syncthreads();
  if (tid < 64)
    s_vcol[tid] = s_vpart[tid * 4] + s_vpart[tid * 4 + 1] +
                  s_vpart[tid * 4 + 2] + s_vpart[tid * 4 + 3];
  __syncthreads();

  // ---- X = acc - lse ⊗ vcol: scatter to LDS, then coalesced b128 stores ----
#pragma unroll
  for (int tj = 0; tj < 2; ++tj)
#pragma unroll
    for (int r = 0; r < 8; ++r) {
      int ql = mrow0 + r + 8 * hl;
      int dl = ncol0 + 16 * tj + nl;
      Sf[ql * 64 + dl] = acc[tj][r] - s_lse[ql] * s_vcol[dl];
    }
  __syncthreads();
  {
    int row = tid >> 2, off = (tid & 3) * 16;
    float* dst = &X[(size_t)(b * kS + qt * 64 + row) * kD + h * kDk + off];
#pragma unroll
    for (int c = 0; c < 4; ++c)
      *(float4*)&dst[4 * c] = *(float4*)&Sf[row * 64 + off + 4 * c];
  }
}

// ---------------------------------------------------------------------------
extern "C" void kernel_launch(void* const* d_in, const int* in_sizes, int n_in,
                              void* d_out, int out_size, void* d_ws,
                              size_t ws_size, hipStream_t stream) {
  (void)in_sizes; (void)n_in; (void)out_size; (void)ws_size;
  const float* query = (const float*)d_in[0];
  const float* key_i = (const float*)d_in[1];
  const float* value = (const float*)d_in[2];
  const int*   mask  = (const int*)d_in[3];
  const float* Wq = (const float*)d_in[4];  const float* bq = (const float*)d_in[5];
  const float* Wk = (const float*)d_in[6];  const float* bk = (const float*)d_in[7];
  const float* Wv = (const float*)d_in[8];  const float* bv = (const float*)d_in[9];
  const float* Wo = (const float*)d_in[10]; const float* bo = (const float*)d_in[11];

  char* ws = (char*)d_ws;
  const size_t qkv_bytes = (size_t)kB * kH * kS * kDk * sizeof(unsigned short);  // 8 MB
  unsigned short* qb = (unsigned short*)(ws);
  unsigned short* kb = (unsigned short*)(ws + qkv_bytes);
  unsigned short* vb = (unsigned short*)(ws + 2 * qkv_bytes);  // [B,H,64,S]
  float* X = (float*)(ws + 3 * qkv_bytes);  // 16 MB fp32 [4096,1024]
  float* out = (float*)d_out;

  dim3 blk(256);
  dim3 gp(kD / 64, (kB * kS) / 128);  // 16 x 32 blocks

  proj_gemm_kernel<0><<<gp, blk, 0, stream>>>(query, Wq, bq, qb, nullptr);
  proj_gemm_kernel<0><<<gp, blk, 0, stream>>>(key_i, Wk, bk, kb, nullptr);
  proj_gemm_kernel<2><<<gp, blk, 0, stream>>>(value, Wv, bv, vb, nullptr);

  attn_kernel<<<dim3(kS / 64, kH, kB), blk, 0, stream>>>(qb, kb, vb, mask, X);

  proj_gemm_kernel<1><<<gp, blk, 0, stream>>>(X, Wo, bo, nullptr, out);
}
